// CrossAssetGNN_18433999635191
// MI455X (gfx1250) — matrix operand order repaired and checked by reading, not compiled, not run on hardware
//
#include <hip/hip_runtime.h>
#include <hip/hip_bf16.h>

#define B_   32
#define A_   128
#define N_   130
#define T_   128
#define DIN_ 64
#define H_   128
#define E_   16770          // N*(N-1)
#define BN_  (B_*N_)        // 4160
#define P0_  15             // embedding positions needed
#define P1_  7              // conv1 positions
#define P2_  3              // conv2 positions

typedef __attribute__((ext_vector_type(16))) _Float16 v16h;
typedef __attribute__((ext_vector_type(8)))  _Float16 v8h;
typedef __attribute__((ext_vector_type(4)))  _Float16 v4h;
typedef __attribute__((ext_vector_type(8)))  float    v8f;

// ---------------------------------------------------------------- WMMA core
__device__ __forceinline__ v8f wmma16(v16h a, v16h b, v8f c) {
  return __builtin_amdgcn_wmma_f32_16x16x32_f16(
      false, a, false, b, (short)0, c, false, false);
}

// A fragment: 16(M) x 32(K) row-major tile with leading dim ld (halves).
// Lane layout (ISA): lanes 0-15 -> M=lane, K {klo..klo+7, 16+klo..16+klo+7}
// with klo = 8*(lane>=16).  Both runs are contiguous -> two b128 LDS loads.
__device__ __forceinline__ v16h load_a16(const _Float16* p, int ld, int lane) {
  int m   = lane & 15;
  int klo = (lane >> 4) << 3;
  const _Float16* r = p + m * ld + klo;
  v8h lo = *(const v8h*)r;
  v8h hi = *(const v8h*)(r + 16);
  return __builtin_shufflevector(lo, hi, 0, 1, 2, 3, 4, 5, 6, 7,
                                 8, 9, 10, 11, 12, 13, 14, 15);
}

// B fragment from fragment-packed weights: lane's 16 halves are contiguous.
__device__ __forceinline__ v16h load_b_pk(const _Float16* tile, int lane) {
  return *(const v16h*)(tile + lane * 16);
}

// Map packed index -> (k, n) of a [K][N] B matrix.
// packed[( (kt*ntN + nt)*32 + lane )*16 + i] = src[kt*32 + 8*(lane>>4) + (i&7) + 16*(i>>3)][nt*16 + (lane&15)]
__device__ __forceinline__ void unpack_idx(int idx, int ntN, int& k, int& n) {
  int i    = idx & 15;
  int lane = (idx >> 4) & 31;
  int tl   = idx >> 9;
  int nt   = tl % ntN;
  int kt   = tl / ntN;
  k = kt * 32 + ((lane >> 4) << 3) + (i & 7) + ((i >> 3) << 4);
  n = nt * 16 + (lane & 15);
}

// ---------------------------------------------------------------- prepack
// Convert all GEMM weights to f16 *fragment-packed* B layout once.
__global__ void prepack_kernel(const float* W_emb, const float* conv_w,
                               const float* gat_W, const float* ew_W1,
                               const float* head_W1,
                               _Float16* wembPK, _Float16* convPK,
                               _Float16* gatPK, _Float16* ewPK,
                               _Float16* headPK) {
  int tid = blockIdx.x * blockDim.x + threadIdx.x;
  int nt  = gridDim.x * blockDim.x;
  int k, n;
  for (int i = tid; i < DIN_ * H_; i += nt) {           // K=64, N=128
    unpack_idx(i, 8, k, n);
    wembPK[i] = (_Float16)W_emb[n * DIN_ + k];          // src[d=k][h=n]
  }
  for (int i = tid; i < 9 * H_ * H_; i += nt) {         // per (l,tap): K=N=128
    int blk = i / (H_ * H_);
    int l = blk / 3, tap = blk % 3;
    unpack_idx(i % (H_ * H_), 8, k, n);                 // cin=k, cout=n
    convPK[i] = (_Float16)conv_w[((l * H_ + n) * H_ + k) * 3 + tap];
  }
  for (int i = tid; i < 3 * H_ * H_; i += nt) {         // per l: K=N=128
    int l = i / (H_ * H_);
    unpack_idx(i % (H_ * H_), 8, k, n);                 // f=k, o=n
    gatPK[i] = (_Float16)gat_W[(l * H_ + n) * H_ + k];
  }
  for (int i = tid; i < 256 * H_; i += nt) {            // K=256, N=128
    unpack_idx(i, 8, k, n);                             // f=k, h=n
    ewPK[i] = (_Float16)ew_W1[n * 256 + k];
  }
  for (int i = tid; i < A_ * H_ * 64; i += nt) {        // per a: K=128, N=64
    int a = i / (H_ * 64);
    unpack_idx(i % (H_ * 64), 4, k, n);                 // h=k, k64=n
    headPK[i] = (_Float16)head_W1[(a * 64 + n) * H_ + k];
  }
  __builtin_amdgcn_s_wait_tensorcnt(0);   // CDNA5 tensor-counter fence
}

// ---------------------------------------------------------------- embedding
// embH[bn][p][h], p=0..14, t = T-1-p.  (16 rows) x (K=64) x (N=128)
__global__ void emb_kernel(const float* x, const _Float16* wembPK,
                           const float* b_emb, _Float16* embH) {
  __shared__ __align__(16) _Float16 sA[16][DIN_];
  int tid = threadIdx.x, lane = tid & 31, wave = tid >> 5;
  int row0 = blockIdx.x * 16;
  {                                      // stage A: one v4 chunk per thread
    int m = tid >> 4, d = (tid & 15) << 2;
    int r = row0 + m;
    int bn = r / P0_;
    int t = T_ - 1 - (r % P0_);
    const float* src = x + ((long)bn * T_ + t) * DIN_ + d;
    v4h h4;
    h4[0] = (_Float16)src[0]; h4[1] = (_Float16)src[1];
    h4[2] = (_Float16)src[2]; h4[3] = (_Float16)src[3];
    *(v4h*)&sA[m][d] = h4;
  }
  __syncthreads();
  v8f acc = {};
#pragma unroll
  for (int kt = 0; kt < 2; ++kt)
    acc = wmma16(load_a16(&sA[0][kt * 32], DIN_, lane),
                 load_b_pk(wembPK + ((long)(kt * 8 + wave) * 32) * 16, lane),
                 acc);
  int n0 = wave * 16 + (lane & 15);
  int mbase = (lane >> 4) << 3;
  float bias = b_emb[n0];
#pragma unroll
  for (int v = 0; v < 8; ++v) {
    long r = row0 + mbase + v;
    embH[r * H_ + n0] = (_Float16)(acc[v] + bias);
  }
}

// ---------------------------------------------------------------- dilated conv
// Positions-only conv: tap k reads input position 2*o + (2-k) for all layers.
// (16 rows) x (K=384) x (N=128), fused bias + BN + exact GELU.
__global__ void conv_kernel(const _Float16* inH, int Pin,
                            const _Float16* convPK_l, const float* conv_b_l,
                            const float* gamma, const float* beta,
                            const float* mean, const float* var,
                            _Float16* outH, int Pout) {
  __shared__ __align__(16) _Float16 sA[16][3 * H_];
  int tid = threadIdx.x, lane = tid & 31, wave = tid >> 5;
  int row0 = blockIdx.x * 16;
  __builtin_prefetch(convPK_l, 0, 3);
#pragma unroll
  for (int it = 0; it < 3; ++it) {       // 768 v8h chunks, 3 per thread
    int i = tid + it * 256;
    int m = i / 48;
    int j = (i % 48) << 3;               // half offset in row (tap-aligned)
    int tap = j >> 7;
    int c = j & 127;
    int r = row0 + m;
    int bn = r / Pout;
    int o = r % Pout;
    int pidx = 2 * o + (2 - tap);
    *(v8h*)&sA[m][j] = *(const v8h*)(inH + ((long)bn * Pin + pidx) * H_ + c);
  }
  __syncthreads();
  v8f acc = {};
#pragma unroll
  for (int tap = 0; tap < 3; ++tap) {
    const _Float16* wl = convPK_l + (long)tap * H_ * H_;
#pragma unroll
    for (int kt = 0; kt < 4; ++kt)
      acc = wmma16(load_a16(&sA[0][tap * 128 + kt * 32], 3 * H_, lane),
                   load_b_pk(wl + ((long)(kt * 8 + wave) * 32) * 16, lane),
                   acc);
  }
  int n0 = wave * 16 + (lane & 15);
  float inv = gamma[n0] * rsqrtf(var[n0] + 1e-5f);
  float mu = mean[n0], bb = beta[n0], cb = conv_b_l[n0];
  int mbase = (lane >> 4) << 3;
#pragma unroll
  for (int v = 0; v < 8; ++v) {
    long r = row0 + mbase + v;
    float val = (acc[v] + cb - mu) * inv + bb;
    val = 0.5f * val * (1.0f + erff(val * 0.70710678f));
    outH[r * H_ + n0] = (_Float16)val;
  }
}

// ---------------------------------------------------------------- edge MLP
__global__ void ew_kernel(const _Float16* featsH, const int* edge_index,
                          const _Float16* ewPK, const float* ew_b1,
                          const float* ew_W2, const float* ew_b2,
                          float* ew_out) {
  __shared__ __align__(16) _Float16 sA[16][256];
  __shared__ float sH[16][H_];
  int tid = threadIdx.x, lane = tid & 31, wave = tid >> 5;
  long row0 = (long)blockIdx.x * 16;
  const int* srcI = edge_index;
  const int* dstI = edge_index + E_;
#pragma unroll
  for (int it = 0; it < 2; ++it) {       // 512 v8h chunks, 2 per thread
    int i = tid + it * 256;
    int m = i >> 5;
    int f = (i & 31) << 3;
    long r = row0 + m;
    int b = (int)(r / E_), e = (int)(r % E_);
    int node = (f < H_) ? srcI[e] : dstI[e];
    *(v8h*)&sA[m][f] =
        *(const v8h*)(featsH + ((long)b * N_ + node) * H_ + (f & (H_ - 1)));
  }
  __syncthreads();
  v8f acc = {};
#pragma unroll
  for (int kt = 0; kt < 8; ++kt)
    acc = wmma16(load_a16(&sA[0][kt * 32], 256, lane),
                 load_b_pk(ewPK + ((long)(kt * 8 + wave) * 32) * 16, lane),
                 acc);
  int n0 = wave * 16 + (lane & 15);
  float b1 = ew_b1[n0];
  int mbase = (lane >> 4) << 3;
#pragma unroll
  for (int v = 0; v < 8; ++v) sH[mbase + v][n0] = fmaxf(acc[v] + b1, 0.0f);
  __syncthreads();
  if (tid < 16) {
    float s = ew_b2[0];
    for (int h = 0; h < H_; ++h) s += sH[tid][h] * ew_W2[h];
    ew_out[row0 + tid] = 1.0f / (1.0f + expf(-s));
  }
}

// ---------------------------------------------------------------- GAT stages
__global__ void gat_gemm_kernel(const _Float16* nfH, const _Float16* gatPK_l,
                                const float* a_src, const float* a_dst,
                                float* hp, float* node_as, float* node_ad) {
  __shared__ __align__(16) _Float16 sA[16][H_];
  __shared__ float sO[16][H_];
  int tid = threadIdx.x, lane = tid & 31, wave = tid >> 5;
  int row0 = blockIdx.x * 16;
  *(v8h*)((_Float16*)sA + tid * 8) =
      *(const v8h*)(nfH + (long)row0 * H_ + tid * 8);
  __syncthreads();
  v8f acc = {};
#pragma unroll
  for (int kt = 0; kt < 4; ++kt)
    acc = wmma16(load_a16(&sA[0][kt * 32], H_, lane),
                 load_b_pk(gatPK_l + ((long)(kt * 8 + wave) * 32) * 16, lane),
                 acc);
  int n0 = wave * 16 + (lane & 15);
  int mbase = (lane >> 4) << 3;
#pragma unroll
  for (int v = 0; v < 8; ++v) {
    sO[mbase + v][n0] = acc[v];
    hp[((long)row0 + mbase + v) * H_ + n0] = acc[v];
  }
  __syncthreads();
  if (tid < 16) {
    float s1 = 0.f, s2 = 0.f;
    for (int h = 0; h < H_; ++h) {
      float v = sO[tid][h];
      s1 += v * a_src[h];
      s2 += v * a_dst[h];
    }
    node_as[row0 + tid] = s1;
    node_ad[row0 + tid] = s2;
  }
}

__global__ void gat_zero_kernel(float* segmax, float* segsum, float* accum) {
  long tid = (long)blockIdx.x * blockDim.x + threadIdx.x;
  long nt = (long)gridDim.x * blockDim.x;
  for (long i = tid; i < BN_; i += nt) { segmax[i] = 0.f; segsum[i] = 0.f; }
  for (long i = tid; i < (long)BN_ * H_; i += nt) accum[i] = 0.f;
}

__global__ void gat_edge1_kernel(const int* edge_index, const float* node_as,
                                 const float* node_ad, const float* ew,
                                 float* alpha, float* segmax) {
  long i = (long)blockIdx.x * blockDim.x + threadIdx.x;
  if (i >= (long)B_ * E_) return;
  int b = (int)(i / E_), e = (int)(i % E_);
  int s = edge_index[e], d = edge_index[E_ + e];
  float v = node_as[b * N_ + s] + node_ad[b * N_ + d];
  v = (v > 0.f) ? v : 0.2f * v;                       // leaky_relu
  v *= ew[i];
  alpha[i] = v;
  // seg-max baseline is 0 (reference inits zeros); positive floats order like
  // their int bit patterns, so integer atomicMax suffices.
  if (v > 0.f) atomicMax((int*)&segmax[b * N_ + d], __float_as_int(v));
}

__global__ void gat_edge2_kernel(const int* edge_index, float* alpha,
                                 const float* segmax, float* segsum) {
  long i = (long)blockIdx.x * blockDim.x + threadIdx.x;
  if (i >= (long)B_ * E_) return;
  int b = (int)(i / E_), e = (int)(i % E_);
  int d = edge_index[E_ + e];
  float v = expf(alpha[i] - segmax[b * N_ + d]);
  alpha[i] = v;
  atomicAdd(&segsum[b * N_ + d], v);
}

__global__ void gat_edge3_kernel(const int* edge_index, const float* alpha,
                                 const float* segsum, const float* hp,
                                 float* accum) {
  long idx = (long)blockIdx.x * 8 + (threadIdx.x >> 5);  // one wave per edge
  if (idx >= (long)B_ * E_) return;
  int lane = threadIdx.x & 31;
  int b = (int)(idx / E_), e = (int)(idx % E_);
  int s = edge_index[e], d = edge_index[E_ + e];
  float coef = alpha[idx] / (segsum[b * N_ + d] + 1e-8f);
  const float* hsrc = hp + ((long)b * N_ + s) * H_;
  float* adst = accum + ((long)b * N_ + d) * H_;
#pragma unroll
  for (int j = 0; j < 4; ++j) {
    int h = lane + 32 * j;
    atomicAdd(&adst[h], hsrc[h] * coef);
  }
}

__global__ void gat_elu_kernel(const float* accum, _Float16* nfH_out) {
  long i = (long)blockIdx.x * blockDim.x + threadIdx.x;
  if (i >= (long)BN_ * H_) return;
  float v = accum[i];
  v = (v > 0.f) ? v : (expf(v) - 1.0f);
  nfH_out[i] = (_Float16)v;
}

// ---------------------------------------------------------------- heads
__global__ void head_kernel(const _Float16* nfH, const _Float16* headPK,
                            const float* head_b1, const float* head_W2,
                            const float* head_b2, float* out) {
  __shared__ __align__(16) _Float16 sA[32][H_];
  __shared__ float sH[32][64];
  __shared__ float sL[32][3];
  int a = blockIdx.x;
  int tid = threadIdx.x, lane = tid & 31, wave = tid >> 5;
#pragma unroll
  for (int it = 0; it < 2; ++it) {       // 512 v8h chunks
    int i = tid + it * 256;
    int bb = i >> 4;
    int c = (i & 15) << 3;
    *(v8h*)&sA[bb][c] = *(const v8h*)(nfH + ((long)bb * N_ + a) * H_ + c);
  }
  __syncthreads();
  int mt = wave >> 2, ntl = wave & 3;    // 2 M-tiles x 4 N-tiles
  const _Float16* wpk = headPK + (long)a * H_ * 64;
  v8f acc = {};
#pragma unroll
  for (int kt = 0; kt < 4; ++kt)
    acc = wmma16(load_a16(&sA[mt * 16][kt * 32], H_, lane),
                 load_b_pk(wpk + ((long)(kt * 4 + ntl) * 32) * 16, lane), acc);
  int n0 = ntl * 16 + (lane & 15);
  float b1 = head_b1[a * 64 + n0];
  int mbase = (lane >> 4) << 3;
#pragma unroll
  for (int v = 0; v < 8; ++v)
    sH[mt * 16 + mbase + v][n0] = fmaxf(acc[v] + b1, 0.f);
  __syncthreads();
  if (tid < 96) {
    int bb = tid / 3, o = tid % 3;
    float s = head_b2[a * 3 + o];
    const float* w2 = head_W2 + (a * 3 + o) * 64;
    for (int k = 0; k < 64; ++k) s += sH[bb][k] * w2[k];
    sL[bb][o] = s;
  }
  __syncthreads();
  if (tid < 32) {
    float l0 = sL[tid][0], l1 = sL[tid][1], l2 = sL[tid][2];
    float mx = fmaxf(l0, fmaxf(l1, l2));
    float e0 = expf(l0 - mx), e1 = expf(l1 - mx), e2 = expf(l2 - mx);
    float inv = 1.f / (e0 + e1 + e2);
    long base = ((long)tid * A_ + a) * 3;
    out[base + 0] = l0; out[base + 1] = l1; out[base + 2] = l2;
    long pb = (long)B_ * A_ * 3 + base;
    out[pb + 0] = e0 * inv; out[pb + 1] = e1 * inv; out[pb + 2] = e2 * inv;
  }
}

// ---------------------------------------------------------------- launch
extern "C" void kernel_launch(void* const* d_in, const int* in_sizes, int n_in,
                              void* d_out, int out_size, void* d_ws,
                              size_t ws_size, hipStream_t stream) {
  (void)in_sizes; (void)n_in; (void)out_size; (void)ws_size;
  const float* x          = (const float*)d_in[0];
  const int*   edge_index = (const int*)d_in[1];
  const float* W_emb      = (const float*)d_in[2];
  const float* b_emb      = (const float*)d_in[3];
  const float* conv_w     = (const float*)d_in[4];
  const float* conv_b     = (const float*)d_in[5];
  const float* bn_gamma   = (const float*)d_in[6];
  const float* bn_beta    = (const float*)d_in[7];
  const float* bn_mean    = (const float*)d_in[8];
  const float* bn_var     = (const float*)d_in[9];
  const float* gat_W      = (const float*)d_in[10];
  const float* gat_a_src  = (const float*)d_in[11];
  const float* gat_a_dst  = (const float*)d_in[12];
  const float* ew_W1      = (const float*)d_in[13];
  const float* ew_b1      = (const float*)d_in[14];
  const float* ew_W2      = (const float*)d_in[15];
  const float* ew_b2      = (const float*)d_in[16];
  const float* head_W1    = (const float*)d_in[17];
  const float* head_b1    = (const float*)d_in[18];
  const float* head_W2    = (const float*)d_in[19];
  const float* head_b2    = (const float*)d_in[20];

  char* ws = (char*)d_ws;
  size_t off = 0;
  auto alloc = [&](size_t bytes) -> void* {
    void* p = ws + off;
    off += (bytes + 255) & ~(size_t)255;
    return p;
  };
  _Float16* wembPK = (_Float16*)alloc((size_t)DIN_ * H_ * 2);
  _Float16* convPK = (_Float16*)alloc((size_t)9 * H_ * H_ * 2);
  _Float16* gatPK  = (_Float16*)alloc((size_t)3 * H_ * H_ * 2);
  _Float16* ewPK   = (_Float16*)alloc((size_t)256 * H_ * 2);
  _Float16* headPK = (_Float16*)alloc((size_t)A_ * H_ * 64 * 2);
  _Float16* embH   = (_Float16*)alloc((size_t)BN_ * P0_ * H_ * 2);
  _Float16* c1H    = (_Float16*)alloc((size_t)BN_ * P1_ * H_ * 2);
  _Float16* c2H    = (_Float16*)alloc((size_t)BN_ * P2_ * H_ * 2);
  _Float16* featsH = (_Float16*)alloc((size_t)BN_ * H_ * 2);
  float* ew_out    = (float*)alloc((size_t)B_ * E_ * 4);
  float* hp        = (float*)alloc((size_t)BN_ * H_ * 4);
  float* node_as   = (float*)alloc((size_t)BN_ * 4);
  float* node_ad   = (float*)alloc((size_t)BN_ * 4);
  float* alpha     = (float*)alloc((size_t)B_ * E_ * 4);
  float* segmax    = (float*)alloc((size_t)BN_ * 4);
  float* segsum    = (float*)alloc((size_t)BN_ * 4);
  float* accum     = (float*)alloc((size_t)BN_ * H_ * 4);
  _Float16* nfH    = (_Float16*)alloc((size_t)BN_ * H_ * 2);

  prepack_kernel<<<256, 256, 0, stream>>>(W_emb, conv_w, gat_W, ew_W1, head_W1,
                                          wembPK, convPK, gatPK, ewPK, headPK);

  emb_kernel<<<(BN_ * P0_) / 16, 256, 0, stream>>>(x, wembPK, b_emb, embH);

  conv_kernel<<<(BN_ * P1_) / 16, 256, 0, stream>>>(
      embH, P0_, convPK + (size_t)0 * 3 * H_ * H_, conv_b + 0 * H_,
      bn_gamma + 0 * H_, bn_beta + 0 * H_, bn_mean + 0 * H_, bn_var + 0 * H_,
      c1H, P1_);
  conv_kernel<<<(BN_ * P2_) / 16, 256, 0, stream>>>(
      c1H, P1_, convPK + (size_t)1 * 3 * H_ * H_, conv_b + 1 * H_,
      bn_gamma + 1 * H_, bn_beta + 1 * H_, bn_mean + 1 * H_, bn_var + 1 * H_,
      c2H, P2_);
  conv_kernel<<<BN_ / 16, 256, 0, stream>>>(
      c2H, P2_, convPK + (size_t)2 * 3 * H_ * H_, conv_b + 2 * H_,
      bn_gamma + 2 * H_, bn_beta + 2 * H_, bn_mean + 2 * H_, bn_var + 2 * H_,
      featsH, 1);

  ew_kernel<<<(B_ * E_) / 16, 256, 0, stream>>>(featsH, edge_index, ewPK,
                                                ew_b1, ew_W2, ew_b2, ew_out);

  long edges = (long)B_ * E_;
  int g_edge = (int)((edges + 255) / 256);
  int g_ed3  = (int)((edges + 7) / 8);
  int g_elu  = (int)(((long)BN_ * H_ + 255) / 256);

  const _Float16* nf_in = featsH;
  for (int i = 0; i < 3; ++i) {
    gat_zero_kernel<<<1024, 256, 0, stream>>>(segmax, segsum, accum);
    gat_gemm_kernel<<<BN_ / 16, 256, 0, stream>>>(
        nf_in, gatPK + (size_t)i * H_ * H_, gat_a_src + i * H_,
        gat_a_dst + i * H_, hp, node_as, node_ad);
    gat_edge1_kernel<<<g_edge, 256, 0, stream>>>(edge_index, node_as, node_ad,
                                                 ew_out, alpha, segmax);
    gat_edge2_kernel<<<g_edge, 256, 0, stream>>>(edge_index, alpha, segmax,
                                                 segsum);
    gat_edge3_kernel<<<g_ed3, 256, 0, stream>>>(edge_index, alpha, segsum, hp,
                                                accum);
    gat_elu_kernel<<<g_elu, 256, 0, stream>>>(accum, nfH);
    nf_in = nfH;
  }

  head_kernel<<<A_, 256, 0, stream>>>(nfH, headPK, head_b1, head_W2, head_b2,
                                      (float*)d_out);
}
